// TaskNet_34179349741586
// MI455X (gfx1250) — matrix operand order repaired and checked by reading, not compiled, not run on hardware
//
#include <hip/hip_runtime.h>
#include <hip/hip_bf16.h>

// ---------------------------------------------------------------------------
// Fused LSTM (B=4096, T=193, D=H=50) + FC(V=11) + log_softmax for gfx1250.
// bf16 WMMA (v_wmma_f32_16x16x32_bf16) for input projection + recurrence,
// fp32 cell math. One block = 16 batch rows, fully independent recurrence.
// Round 2: branch-free uniform 4 tiles/wave (N padded to 256), x_t register
// double-buffering (global latency off the critical path), LDS overlay
// (weights staging aliased with gate buffer), fast exp/rcp nonlinearities,
// 2 barriers per timestep.
// ---------------------------------------------------------------------------

#define BB   4096
#define TT   193
#define DD   50
#define HH   50
#define VV   11
#define G4H  200          // 4*H
#define NT16 16           // gate-axis N tiles (padded: 13 real + 3 zero)
#define NPAD 256          // 16*16
#define KPAD 64           // K padded 50 -> 64 (two K=32 WMMA slabs)
#define ROWS 16           // batch rows per block
#define THREADS 128       // 4 waves, each owns 4 N tiles

typedef __attribute__((ext_vector_type(16))) __bf16 v16bf;
typedef __attribute__((ext_vector_type(8)))  float  v8f;

// ---- LDS overlay (72.8 KB): weight staging dead after preload, alias sG ----
// [0      , 32768)  sWih  ushort[256*64]   (setup only)
// [32768  , 65536)  sWhh  ushort[256*64]   (setup only)
// [0      , 16384)  sG    float [16*256]   (loop; aliases sWih)
// [65536  , 67584)  sX    ushort[16*64]
// [67584  , 69632)  sH    ushort[16*64]
// [69632  , 70656)  sBias float [256]
// [70656  , 73856)  sHf   float [16*50]
// [73856  , 74560)  sLogit float[16*11]
#define SMEM_BYTES 74560

// float -> bf16, round-to-nearest-even
__device__ __forceinline__ unsigned short f2bf(float f) {
    unsigned int u = __float_as_uint(f);
    u += 0x7FFFu + ((u >> 16) & 1u);
    return (unsigned short)(u >> 16);
}

// 16-element bf16 fragment from two 16-byte LDS chunks (2x ds_load_b128)
__device__ __forceinline__ v16bf ld_frag(const unsigned short* p0,
                                         const unsigned short* p1) {
    union { uint4 q[2]; v16bf v; } u;
    u.q[0] = *reinterpret_cast<const uint4*>(p0);
    u.q[1] = *reinterpret_cast<const uint4*>(p1);
    return u.v;
}

__device__ __forceinline__ float fsigmoid(float x) {
    return __builtin_amdgcn_rcpf(1.0f + __expf(-x));
}
__device__ __forceinline__ float ftanh(float x) {
    return 1.0f - 2.0f * __builtin_amdgcn_rcpf(1.0f + __expf(2.0f * x));
}

__global__ __launch_bounds__(THREADS)
void lstm_fused_kernel(const float* __restrict__ x,      // [B,T,D]
                       const float* __restrict__ W_ih,   // [4H,D]
                       const float* __restrict__ W_hh,   // [4H,H]
                       const float* __restrict__ b_ih,   // [4H]
                       const float* __restrict__ b_hh,   // [4H]
                       const float* __restrict__ fc_w,   // [V,H]
                       const float* __restrict__ fc_b,   // [V]
                       float* __restrict__ out)          // [B,V]
{
    __shared__ __align__(16) unsigned char smem[SMEM_BYTES];
    unsigned short* sWih  = (unsigned short*)(smem);
    unsigned short* sWhh  = (unsigned short*)(smem + 32768);
    float*          sG    = (float*)(smem);                 // aliases sWih/sWhh
    unsigned short* sX    = (unsigned short*)(smem + 65536);
    unsigned short* sH    = (unsigned short*)(smem + 67584);
    float*          sBias = (float*)(smem + 69632);
    float*          sHf   = (float*)(smem + 70656);
    float*          sLogit= (float*)(smem + 73856);

    const int tid  = threadIdx.x;
    const int lane = tid & 31;
    const int wave = tid >> 5;
    const int lo   = lane & 15;     // A: row M  / B: column N within tile
    const int hi   = lane >> 4;     // lane-half selector
    const int bq   = blockIdx.x;    // batch tile index (16 rows)

    // ---- one-time setup: zero-pad, convert weights to bf16, bias --------
    {
        unsigned int* z = (unsigned int*)smem;              // sWih+sWhh region
        for (int i = tid; i < 65536 / 4; i += THREADS) z[i] = 0u;
        unsigned int* zx = (unsigned int*)sX;               // sX + sH (contig)
        for (int i = tid; i < 4096 / 4; i += THREADS) zx[i] = 0u;
    }
    __syncthreads();
    for (int i = tid; i < G4H * DD; i += THREADS) {
        int g = i / DD, d = i % DD;
        sWih[g * KPAD + d] = f2bf(W_ih[i]);
        sWhh[g * KPAD + d] = f2bf(W_hh[i]);
    }
    for (int i = tid; i < NPAD; i += THREADS)
        sBias[i] = (i < G4H) ? (b_ih[i] + b_hh[i]) : 0.0f;
    __syncthreads();

    // ---- B-fragment register preload: wave w owns N tiles 4w..4w+3 ------
    v16bf Bih[4][2], Bhh[4][2];
    #pragma unroll
    for (int ti = 0; ti < 4; ++ti) {
        const int tn   = wave * 4 + ti;
        const int rowb = (tn * 16 + lo) * KPAD + hi * 8;    // 8+8 K per lane-half
        #pragma unroll
        for (int kb = 0; kb < 2; ++kb) {
            Bih[ti][kb] = ld_frag(&sWih[rowb + kb * 32], &sWih[rowb + kb * 32 + 16]);
            Bhh[ti][kb] = ld_frag(&sWhh[rowb + kb * 32], &sWhh[rowb + kb * 32 + 16]);
        }
    }

    // ---- stage x_0 into sX (bf16) ---------------------------------------
    for (int i = tid; i < ROWS * DD; i += THREADS) {
        int r = i / DD, d = i % DD;
        sX[r * KPAD + d] = f2bf(x[((size_t)(bq * ROWS + r) * TT) * DD + d]);
    }
    __syncthreads();   // also fences B-frag LDS reads vs. first sG writes

    // ---- cell state in registers (fixed thread->element map) ------------
    float c_state[7];
    #pragma unroll
    for (int e = 0; e < 7; ++e) c_state[e] = 0.0f;

    // ---- time loop: 2 barriers/step, x double-buffered through regs -----
    for (int t = 0; t < TT; ++t) {
        // prefetch x_{t+1} into registers (latency hidden behind WMMA phase)
        const int tn_next = (t + 1 < TT) ? (t + 1) : (TT - 1);
        float xr[7];
        #pragma unroll
        for (int e = 0; e < 7; ++e) {
            const int i = tid + e * THREADS;
            if (i < ROWS * DD) {
                const int r = i / DD, d = i % DD;
                xr[e] = x[((size_t)(bq * ROWS + r) * TT + tn_next) * DD + d];
            }
        }

        // A fragments (shared across this wave's 4 N tiles)
        const int abase = lo * KPAD + hi * 8;   // 8+8 K per lane-half
        v16bf ax0 = ld_frag(&sX[abase],      &sX[abase + 16]);
        v16bf ax1 = ld_frag(&sX[abase + 32], &sX[abase + 48]);
        v16bf ah0 = ld_frag(&sH[abase],      &sH[abase + 16]);
        v16bf ah1 = ld_frag(&sH[abase + 32], &sH[abase + 48]);

        // branch-free WMMA phase: 4 tiles x 4 WMMA, uniform across waves
        #pragma unroll
        for (int ti = 0; ti < 4; ++ti) {
            v8f acc = {0.f, 0.f, 0.f, 0.f, 0.f, 0.f, 0.f, 0.f};
            acc = __builtin_amdgcn_wmma_f32_16x16x32_bf16(
                      false, ax0, false, Bih[ti][0], (short)0, acc, false, false);
            acc = __builtin_amdgcn_wmma_f32_16x16x32_bf16(
                      false, ax1, false, Bih[ti][1], (short)0, acc, false, false);
            acc = __builtin_amdgcn_wmma_f32_16x16x32_bf16(
                      false, ah0, false, Bhh[ti][0], (short)0, acc, false, false);
            acc = __builtin_amdgcn_wmma_f32_16x16x32_bf16(
                      false, ah1, false, Bhh[ti][1], (short)0, acc, false, false);
            const int tn = wave * 4 + ti;
            #pragma unroll
            for (int r = 0; r < 8; ++r)
                sG[(r + hi * 8) * NPAD + tn * 16 + lo] = acc[r];
        }
        __syncthreads();   // gates ready; all sX/sH reads complete

        // elementwise LSTM cell update + stage x_{t+1} (same (r,j) map: D==H)
        #pragma unroll
        for (int e = 0; e < 7; ++e) {
            const int i = tid + e * THREADS;
            if (i < ROWS * HH) {
                const int r = i / HH, j = i % HH;
                const float* gr = &sG[r * NPAD];
                float ig = fsigmoid(gr[j         ] + sBias[j         ]);
                float fg = fsigmoid(gr[j + HH    ] + sBias[j + HH    ]);
                float gg = ftanh   (gr[j + 2*HH  ] + sBias[j + 2*HH  ]);
                float og = fsigmoid(gr[j + 3*HH  ] + sBias[j + 3*HH  ]);
                float c  = fg * c_state[e] + ig * gg;
                c_state[e] = c;
                float h = og * ftanh(c);
                sH[r * KPAD + j] = f2bf(h);
                sHf[r * HH + j]  = h;
                sX[r * KPAD + j] = f2bf(xr[e]);   // stage next timestep
            }
        }
        __syncthreads();   // sH + sX ready for next step
    }

    // ---- epilogue: logits = relu(h) @ fc_w^T + fc_b ---------------------
    for (int p = tid; p < ROWS * VV; p += THREADS) {
        const int r = p / VV, v = p % VV;
        float acc = fc_b[v];
        #pragma unroll 10
        for (int k = 0; k < HH; ++k) {
            float hk = sHf[r * HH + k];
            hk = hk > 0.0f ? hk : 0.0f;
            acc += hk * fc_w[v * HH + k];
        }
        sLogit[r * VV + v] = acc;
    }
    __syncthreads();

    // log_softmax per row
    if (tid < ROWS) {
        const int r = tid;
        float m = -1e30f;
        #pragma unroll
        for (int v = 0; v < VV; ++v) m = fmaxf(m, sLogit[r * VV + v]);
        float s = 0.0f;
        #pragma unroll
        for (int v = 0; v < VV; ++v) s += __expf(sLogit[r * VV + v] - m);
        const float ls = __logf(s);
        #pragma unroll
        for (int v = 0; v < VV; ++v)
            out[(size_t)(bq * ROWS + r) * VV + v] = sLogit[r * VV + v] - m - ls;
    }
}

extern "C" void kernel_launch(void* const* d_in, const int* in_sizes, int n_in,
                              void* d_out, int out_size, void* d_ws, size_t ws_size,
                              hipStream_t stream) {
    const float* x    = (const float*)d_in[0];
    const float* W_ih = (const float*)d_in[1];
    const float* W_hh = (const float*)d_in[2];
    const float* b_ih = (const float*)d_in[3];
    const float* b_hh = (const float*)d_in[4];
    const float* fc_w = (const float*)d_in[5];
    const float* fc_b = (const float*)d_in[6];
    float* out = (float*)d_out;

    dim3 grid(BB / ROWS);   // 256 blocks, each owns 16 batch rows end-to-end
    dim3 block(THREADS);
    lstm_fused_kernel<<<grid, block, 0, stream>>>(x, W_ih, W_hh, b_ih, b_hh,
                                                  fc_w, fc_b, out);
}
